// FNetLayer_11166914970281
// MI455X (gfx1250) — matrix operand order repaired and checked by reading, not compiled, not run on hardware
//
#include <hip/hip_runtime.h>
#include <math.h>

typedef __bf16 bf16;
typedef __attribute__((ext_vector_type(16))) __bf16 v16bf;
typedef __attribute__((ext_vector_type(8)))  __bf16 v8bf;
typedef __attribute__((ext_vector_type(4)))  __bf16 v4bf;
typedef __attribute__((ext_vector_type(8)))  float  v8f;
typedef __attribute__((ext_vector_type(4)))  float  v4f;

// ---------------------------------------------------------------------------
// helpers
// ---------------------------------------------------------------------------
static __device__ inline v16bf frag_combine(v8bf lo, v8bf hi) {
    return __builtin_shufflevector(lo, hi, 0,1,2,3,4,5,6,7,8,9,10,11,12,13,14,15);
}

// ---------------------------------------------------------------------------
// element-wise / generator kernels
// ---------------------------------------------------------------------------
__global__ __launch_bounds__(256) void cvt_f32_to_bf16_k(const float* __restrict__ in,
                                                         bf16* __restrict__ out, long long n) {
    long long i = (long long)blockIdx.x * 256 + threadIdx.x;
    if (i < n) out[i] = (bf16)in[i];
}

// hidden DFT matrices (1024x1024, symmetric) generated directly in B-fragment
// packed layout: [K/32][N/16][lane(32)][16 elems]
__global__ __launch_bounds__(256) void gen_hidden_dft_k(bf16* __restrict__ Ch,
                                                        bf16* __restrict__ Sh) {
    long long p = (long long)blockIdx.x * 256 + threadIdx.x;
    if (p >= 1024LL * 1024LL) return;
    int e    = (int)(p & 15);
    int lane = (int)((p >> 4) & 31);
    int ntl  = (int)((p >> 9) & 63);      // N/16 = 64
    int kt   = (int)(p >> 15);
    int k = kt * 32 + (lane >> 4) * 16 + e;
    int n = ntl * 16 + (lane & 15);
    int ph = (k * n) & 1023;
    float ang = (float)ph * (6.283185307179586f / 1024.0f);
    Ch[p] = (bf16)cosf(ang);
    Sh[p] = (bf16)sinf(ang);
}

// sequence DFT operator [C_S | -S_S], row-major 2048 x 4096 bf16 (A operand)
__global__ __launch_bounds__(256) void gen_seq_dft_k(bf16* __restrict__ Cs) {
    long long i = (long long)blockIdx.x * 256 + threadIdx.x;
    if (i >= 2048LL * 4096LL) return;
    int sp = (int)(i >> 12);
    int k  = (int)(i & 4095);
    float v;
    if (k < 2048) {
        int ph = (sp * k) & 2047;
        v = cosf((float)ph * (6.283185307179586f / 2048.0f));
    } else {
        int ph = (sp * (k - 2048)) & 2047;
        v = -sinf((float)ph * (6.283185307179586f / 2048.0f));
    }
    Cs[i] = (bf16)v;
}

// pack a row-major f32 K x N matrix into bf16 B-fragment layout
__global__ __launch_bounds__(256) void pack_b_k(const float* __restrict__ in,
                                                bf16* __restrict__ out, int K, int N) {
    long long p = (long long)blockIdx.x * 256 + threadIdx.x;
    long long total = (long long)K * N;
    if (p >= total) return;
    int NT   = N >> 4;
    int e    = (int)(p & 15);
    int lane = (int)((p >> 4) & 31);
    long long t = p >> 9;
    int ntl = (int)(t % NT);
    int kt  = (int)(t / NT);
    int k = kt * 32 + (lane >> 4) * 16 + e;
    int n = ntl * 16 + (lane & 15);
    out[p] = (bf16)in[(size_t)k * N + n];
}

// ---------------------------------------------------------------------------
// WMMA GEMM: C = A(row-major bf16, MxK) * B(packed bf16, KxN), block 128x128
// Staging uses CDNA5 async global->LDS DMA (ASYNCcnt), double buffered.
// ---------------------------------------------------------------------------
enum { EPI_PACKB = 0, EPI_ADDX = 1, EPI_GELU = 2, EPI_RESBIAS = 3 };

template <int EPI>
__global__ __launch_bounds__(256) void gemm_bf16_wmma(
    const bf16* __restrict__ A, const bf16* __restrict__ Bp,
    int M, int N, int K,
    const float* __restrict__ res, const float* __restrict__ bias,
    float* __restrict__ outF, bf16* __restrict__ outB,
    int kOffsetPack,
    long long sA, long long sB, long long sRes, long long sOut) {

    const int z = blockIdx.z;
    A  += (long long)z * sA;
    Bp += (long long)z * sB;
    if (res)  res  += (long long)z * sRes;
    if (outF) outF += (long long)z * sOut;

    const int tid  = threadIdx.x;
    const int wave = tid >> 5;
    const int lane = tid & 31;
    const int wM   = wave >> 2;       // 0..1  (64 rows each)
    const int wN   = wave & 3;        // 0..3  (32 cols each)
    const int m0   = blockIdx.y * 128;
    const int n0   = blockIdx.x * 128;

    __shared__ bf16  lsA[2][4096];    // double-buffered: 8 subtiles x 32 lanes x 16
    __shared__ bf16  lsB[2][4096];
    __shared__ float lsC[2048];       // 8 waves x 256 (epilogue transpose)

    v8f acc[4][2] = {};

    // --- A staging plan (swizzle into fragment layout) ---
    const int rowA = tid >> 1;        // 0..127
    const int half = tid & 1;         // kk base = half*16
    const int stA  = rowA >> 4;
    const int liA  = rowA & 15;
    const size_t aBaseElem = (size_t)(m0 + rowA) * K + half * 16;
    const int dA0 = (stA * 32 + half * 16 + liA) * 16;       // lane group half, elems 0..7
    const int dA1 = (stA * 32 + half * 16 + liA) * 16 + 8;   // lane group half, elems 8..15

    const int NT = N >> 4;
    const size_t bTileElem = (size_t)(n0 >> 4) * 512 + (size_t)tid * 16;

    const unsigned long long aBase64 = (unsigned long long)A;
    const unsigned long long bBase64 = (unsigned long long)Bp;

    // async global->LDS staging of one 128x32 A tile + 32x128 packed B tile
    auto stage = [&](int kt, int buf) {
        unsigned la0 = (unsigned)(unsigned long long)&lsA[buf][dA0];
        unsigned la1 = (unsigned)(unsigned long long)&lsA[buf][dA1];
        unsigned lb0 = (unsigned)(unsigned long long)&lsB[buf][tid * 16];
        unsigned va0 = (unsigned)((aBaseElem + (size_t)kt * 32) * 2);
        unsigned va1 = va0 + 16;
        unsigned vb0 = (unsigned)(((size_t)kt * NT * 512 + bTileElem) * 2);
        asm volatile("global_load_async_to_lds_b128 %0, %1, %2"
                     :: "v"(la0), "v"(va0), "s"(aBase64) : "memory");
        asm volatile("global_load_async_to_lds_b128 %0, %1, %2"
                     :: "v"(la1), "v"(va1), "s"(aBase64) : "memory");
        asm volatile("global_load_async_to_lds_b128 %0, %1, %2"
                     :: "v"(lb0), "v"(vb0), "s"(bBase64) : "memory");
        // INST_OFFSET applies to both global and LDS address -> second 16B chunk
        asm volatile("global_load_async_to_lds_b128 %0, %1, %2 offset:16"
                     :: "v"(lb0), "v"(vb0), "s"(bBase64) : "memory");
    };

    const int KT = K >> 5;
    stage(0, 0);
    for (int kt = 0; kt < KT; ++kt) {
        const int cur = kt & 1;
        asm volatile("s_wait_asynccnt 0x0" ::: "memory");
        __syncthreads();                       // prefetched buffer visible to all waves
        if (kt + 1 < KT) stage(kt + 1, cur ^ 1);

        v16bf af[4], bfr[2];
#pragma unroll
        for (int mt = 0; mt < 4; ++mt) {
            int st = wM * 4 + mt;
            v8bf lo = *(v8bf*)&lsA[cur][(st * 32 + lane) * 16];
            v8bf hi = *(v8bf*)&lsA[cur][(st * 32 + lane) * 16 + 8];
            af[mt] = frag_combine(lo, hi);
        }
#pragma unroll
        for (int nt = 0; nt < 2; ++nt) {
            int st = wN * 2 + nt;
            v8bf lo = *(v8bf*)&lsB[cur][(st * 32 + lane) * 16];
            v8bf hi = *(v8bf*)&lsB[cur][(st * 32 + lane) * 16 + 8];
            bfr[nt] = frag_combine(lo, hi);
        }
#pragma unroll
        for (int mt = 0; mt < 4; ++mt)
#pragma unroll
            for (int nt = 0; nt < 2; ++nt)
                acc[mt][nt] = __builtin_amdgcn_wmma_f32_16x16x32_bf16(
                    false, af[mt], false, bfr[nt], (short)0, acc[mt][nt], false, false);
    }

    // ------------------------------ epilogues ------------------------------
    if constexpr (EPI == EPI_PACKB) {
        // write bf16 result into B-fragment packed AB buffer (next-GEMM K = 4096)
        const int g  = lane >> 4;
        const int nn = lane & 15;
#pragma unroll
        for (int mt = 0; mt < 4; ++mt) {
#pragma unroll
            for (int nt = 0; nt < 2; ++nt) {
                int mTile = m0 + wM * 64 + mt * 16;         // global token row
                int batch = mTile >> 11;                    // /2048
                int sLoc  = mTile & 2047;
                int kBase = kOffsetPack + sLoc;             // 16-aligned
                int ktN   = kBase >> 5;
                int grp   = (kBase >> 4) & 1;
                int nTile = n0 + wN * 32 + nt * 16;
                size_t off = (((size_t)ktN * 64 + (nTile >> 4)) * 32 + (nn + 16 * grp)) * 16 + 8 * g;
                v8bf pk;
#pragma unroll
                for (int r = 0; r < 8; ++r) pk[r] = (bf16)acc[mt][nt][r];
                *(v8bf*)(outB + (size_t)batch * (4096 * 1024) + off) = pk;
            }
        }
    } else {
        float* myC = &lsC[wave * 256];
        const int g  = lane >> 4;
        const int nn = lane & 15;
        const int rr = lane >> 1;
        const int cb = (lane & 1) * 8;
#pragma unroll
        for (int mt = 0; mt < 4; ++mt) {
#pragma unroll
            for (int nt = 0; nt < 2; ++nt) {
                // C fragment -> per-wave LDS tile (row-major 16x16)
#pragma unroll
                for (int r = 0; r < 8; ++r)
                    myC[(r + 8 * g) * 16 + nn] = acc[mt][nt][r];
                asm volatile("s_wait_dscnt 0x0" ::: "memory");
                v4f lo = *(v4f*)&myC[rr * 16 + cb];
                v4f hi = *(v4f*)&myC[rr * 16 + cb + 4];
                asm volatile("s_wait_dscnt 0x0" ::: "memory");

                size_t row = (size_t)(m0 + wM * 64 + mt * 16 + rr);
                size_t col = (size_t)(n0 + wN * 32 + nt * 16 + cb);

                if constexpr (EPI == EPI_ADDX) {
                    const float* rp = res + row * N + col;
                    v4f x0 = *(const v4f*)rp;
                    v4f x1 = *(const v4f*)(rp + 4);
                    float* op = outF + row * N + col;
                    *(v4f*)op       = lo + x0;
                    *(v4f*)(op + 4) = hi + x1;
                } else if constexpr (EPI == EPI_GELU) {
                    v8bf ob;
#pragma unroll
                    for (int j = 0; j < 8; ++j) {
                        float c = (j < 4 ? lo[j] : hi[j - 4]) + bias[col + j];
                        float gv = 0.5f * c * (1.0f + erff(c * 0.70710678118654752f));
                        ob[j] = (bf16)gv;
                    }
                    *(v8bf*)(outB + row * N + col) = ob;
                } else { // EPI_RESBIAS (in-place f32: z = u + acc + b2)
                    const float* rp = res + row * N + col;
                    v4f u0 = *(const v4f*)rp;
                    v4f u1 = *(const v4f*)(rp + 4);
                    float* op = outF + row * N + col;
                    v4f o0, o1;
#pragma unroll
                    for (int j = 0; j < 4; ++j) {
                        o0[j] = u0[j] + lo[j] + bias[col + j];
                        o1[j] = u1[j] + hi[j] + bias[col + 4 + j];
                    }
                    *(v4f*)op       = o0;
                    *(v4f*)(op + 4) = o1;
                }
            }
        }
    }
}

// ---------------------------------------------------------------------------
// LayerNorm over H=1024, one row per 256-thread block (wave32 reductions)
// ---------------------------------------------------------------------------
__global__ __launch_bounds__(256) void layernorm_k(const float* __restrict__ in,
                                                   const float* __restrict__ w,
                                                   const float* __restrict__ b,
                                                   float* __restrict__ outF,
                                                   bf16* __restrict__ outB) {
    const int H = 1024;
    const int row = blockIdx.x;
    const int tid = threadIdx.x;
    v4f v = *(const v4f*)(in + (size_t)row * H + tid * 4);
    float s1 = v[0] + v[1] + v[2] + v[3];
    float s2 = v[0] * v[0] + v[1] * v[1] + v[2] * v[2] + v[3] * v[3];
#pragma unroll
    for (int off = 16; off > 0; off >>= 1) {
        s1 += __shfl_xor(s1, off, 32);
        s2 += __shfl_xor(s2, off, 32);
    }
    __shared__ float r1[8], r2[8];
    if ((tid & 31) == 0) { r1[tid >> 5] = s1; r2[tid >> 5] = s2; }
    __syncthreads();
    s1 = 0.f; s2 = 0.f;
#pragma unroll
    for (int i = 0; i < 8; ++i) { s1 += r1[i]; s2 += r2[i]; }
    float mu   = s1 * (1.0f / H);
    float var  = s2 * (1.0f / H) - mu * mu;
    float rstd = rsqrtf(var + 1e-5f);
    v4f wv = *(const v4f*)(w + tid * 4);
    v4f bv = *(const v4f*)(b + tid * 4);
    v4f o;
#pragma unroll
    for (int j = 0; j < 4; ++j) o[j] = (v[j] - mu) * rstd * wv[j] + bv[j];
    *(v4f*)(outF + (size_t)row * H + tid * 4) = o;
    if (outB) {
        v4bf ob;
#pragma unroll
        for (int j = 0; j < 4; ++j) ob[j] = (bf16)o[j];
        *(v4bf*)(outB + (size_t)row * H + tid * 4) = ob;
    }
}

// ---------------------------------------------------------------------------
// launch
// ---------------------------------------------------------------------------
extern "C" void kernel_launch(void* const* d_in, const int* in_sizes, int n_in,
                              void* d_out, int out_size, void* d_ws, size_t ws_size,
                              hipStream_t stream) {
    const float* x    = (const float*)d_in[0];
    const float* ln1w = (const float*)d_in[1];
    const float* ln1b = (const float*)d_in[2];
    const float* ln2w = (const float*)d_in[3];
    const float* ln2b = (const float*)d_in[4];
    const float* w1   = (const float*)d_in[5];
    const float* b1   = (const float*)d_in[6];
    const float* w2   = (const float*)d_in[7];
    const float* b2   = (const float*)d_in[8];

    const long long Mtok = 8192;      // 4 * 2048
    char* wp = (char*)d_ws;
    auto alloc = [&](size_t bytes) -> void* {
        void* p = (void*)wp;
        wp += (bytes + 255) & ~(size_t)255;
        return p;
    };
    bf16*  xb  = (bf16*)alloc(Mtok * 1024 * 2);          // x as bf16 row-major
    bf16*  Chb = (bf16*)alloc(1024LL * 1024 * 2);        // packed cos DFT (hidden)
    bf16*  Shb = (bf16*)alloc(1024LL * 1024 * 2);        // packed sin DFT (hidden)
    bf16*  Csm = (bf16*)alloc(2048LL * 4096 * 2);        // [C_S | -S_S] row-major
    bf16*  w1p = (bf16*)alloc(1024LL * 4096 * 2);        // packed w1
    bf16*  w2p = (bf16*)alloc(4096LL * 1024 * 2);        // packed w2
    bf16*  AB  = (bf16*)alloc(4LL * 4096 * 1024 * 2);    // packed per-batch [A;B]
    float* y   = (float*)alloc(Mtok * 1024 * 4);         // f32 activations
    bf16*  ub  = (bf16*)alloc(Mtok * 1024 * 2);          // LN1 output bf16
    bf16*  hbf = (bf16*)alloc(Mtok * 4096 * 2);          // GELU hidden bf16

    // 1) precompute / pack
    {
        long long n = Mtok * 1024;
        cvt_f32_to_bf16_k<<<(unsigned)((n + 255) / 256), 256, 0, stream>>>(x, xb, n);
        gen_hidden_dft_k<<<(unsigned)((1024LL * 1024 + 255) / 256), 256, 0, stream>>>(Chb, Shb);
        gen_seq_dft_k<<<(unsigned)((2048LL * 4096 + 255) / 256), 256, 0, stream>>>(Csm);
        pack_b_k<<<(unsigned)((1024LL * 4096 + 255) / 256), 256, 0, stream>>>(w1, w1p, 1024, 4096);
        pack_b_k<<<(unsigned)((4096LL * 1024 + 255) / 256), 256, 0, stream>>>(w2, w2p, 4096, 1024);
    }

    // 2) hidden-dim DFT: A = x*C_H -> AB[k=0..2047], B = x*S_H -> AB[k=2048..4095]
    {
        dim3 g(1024 / 128, 8192 / 128, 1);
        gemm_bf16_wmma<EPI_PACKB><<<g, 256, 0, stream>>>(
            xb, Chb, 8192, 1024, 1024, nullptr, nullptr, nullptr, AB, 0, 0, 0, 0, 0);
        gemm_bf16_wmma<EPI_PACKB><<<g, 256, 0, stream>>>(
            xb, Shb, 8192, 1024, 1024, nullptr, nullptr, nullptr, AB, 2048, 0, 0, 0, 0);
    }

    // 3) sequence-dim DFT + residual: y = x + [C_S|-S_S] @ [A;B]   (per batch)
    {
        dim3 g(1024 / 128, 2048 / 128, 4);
        gemm_bf16_wmma<EPI_ADDX><<<g, 256, 0, stream>>>(
            Csm, AB, 2048, 1024, 4096, x, nullptr, y, nullptr, 0,
            /*sA*/ 0, /*sB*/ 4096LL * 1024, /*sRes*/ 2048LL * 1024, /*sOut*/ 2048LL * 1024);
    }

    // 4) LN1 (in place on y, also emits bf16 operand)
    layernorm_k<<<8192, 256, 0, stream>>>(y, ln1w, ln1b, y, ub);

    // 5) FFN up: h = gelu(u @ w1 + b1)
    {
        dim3 g(4096 / 128, 8192 / 128, 1);
        gemm_bf16_wmma<EPI_GELU><<<g, 256, 0, stream>>>(
            ub, w1p, 8192, 4096, 1024, nullptr, b1, nullptr, hbf, 0, 0, 0, 0, 0);
    }

    // 6) FFN down + residual: y = u + h @ w2 + b2   (in place on y)
    {
        dim3 g(1024 / 128, 8192 / 128, 1);
        gemm_bf16_wmma<EPI_RESBIAS><<<g, 256, 0, stream>>>(
            hbf, w2p, 8192, 1024, 4096, y, b2, y, nullptr, 0, 0, 0, 0, 0);
    }

    // 7) LN2 -> output
    layernorm_k<<<8192, 256, 0, stream>>>(y, ln2w, ln2b, (float*)d_out, nullptr);
}
